// NIQE_76184129896659
// MI455X (gfx1250) — matrix operand order repaired and checked by reading, not compiled, not run on hardware
//
#include <hip/hip_runtime.h>

typedef __attribute__((ext_vector_type(16))) _Float16 v16h;
typedef __attribute__((ext_vector_type(8)))  float    v8f;
typedef __attribute__((ext_vector_type(4)))  unsigned int u32x4;
typedef __attribute__((ext_vector_type(8)))  int      i32x8;
typedef __attribute__((ext_vector_type(4)))  int      i32x4;

#define DIV_UP(a,b) (((a)+(b)-1)/(b))

__device__ __forceinline__ int iclamp(int v,int lo,int hi){ return v<lo?lo:(v>hi?hi:v); }

// ---------------- 1. RGB -> Y (rounded), crop to 1056 x 1920 ----------------
__global__ void niqe_y_kernel(const float* __restrict__ x, float* __restrict__ img)
{
  const int H=1056, W=1920, HS=1080;
  int idx = blockIdx.x*blockDim.x + threadIdx.x;
  int total = 8*H*W;
  if (idx >= total) return;
  int w = idx % W; int t = idx / W; int h = t % H; int b = t / H;
  size_t base = ((size_t)b*3*HS + h)*(size_t)W + w;    // channel stride = HS*W
  float r  = x[base];
  float g  = x[base + (size_t)HS*W];
  float bl = x[base + (size_t)2*HS*W];
  float y = 65.481f*r + 128.553f*g + 24.966f*bl + 16.f;
  img[idx] = rintf(y);                                  // round-half-even, matches jnp.round
}

// ------------- 2. local mean/std normalization (separable 7x7 gauss) -------------
__global__ void niqe_musigma_kernel(const float* __restrict__ img, float* __restrict__ nrm,
                                    int H, int W)
{
  __shared__ float t[22][23];
  __shared__ float h1[22][16];
  __shared__ float h2[22][16];
  int tx = threadIdx.x, ty = threadIdx.y;
  int bx = blockIdx.x*16, by = blockIdx.y*16, b = blockIdx.z;
  const float* im = img + (size_t)b*H*W;
  int tid = ty*16 + tx;
  for (int i = tid; i < 22*22; i += 256) {
    int ly = i/22, lx = i%22;
    int gy = iclamp(by-3+ly, 0, H-1);
    int gx = iclamp(bx-3+lx, 0, W-1);
    t[ly][lx] = im[(size_t)gy*W + gx];
  }
  // normalized separable gaussian, sigma = 7/6
  float g[7]; float gs = 0.f;
  #pragma unroll
  for (int i = 0; i < 7; i++) { float a=(float)(i-3); g[i]=expf(-a*a/(2.f*(7.f/6.f)*(7.f/6.f))); gs+=g[i]; }
  #pragma unroll
  for (int i = 0; i < 7; i++) g[i] /= gs;
  __syncthreads();
  // horizontal pass: s1 = sum g*p, s2 = sum g*p^2
  for (int i = tid; i < 22*16; i += 256) {
    int r = i/16, c = i%16;
    float s1 = 0.f, s2 = 0.f;
    #pragma unroll
    for (int d = 0; d < 7; d++) { float p = t[r][c+d]; s1 += g[d]*p; s2 += g[d]*p*p; }
    h1[r][c] = s1; h2[r][c] = s2;
  }
  __syncthreads();
  int ox = bx+tx, oy = by+ty;
  if (ox >= W || oy >= H) return;
  float mu = 0.f, m2 = 0.f;
  #pragma unroll
  for (int d = 0; d < 7; d++) { mu += g[d]*h1[ty+d][tx]; m2 += g[d]*h2[ty+d][tx]; }
  float sigma = sqrtf(fabsf(m2 - mu*mu));
  float p0 = t[ty+3][tx+3];
  nrm[(size_t)b*H*W + (size_t)oy*W + ox] = (p0 - mu)/(sigma + 1.f);
}

// ---------------- 3. half-size bicubic resize (8-tap, stride 2, symmetric pad) ----------------
__device__ __forceinline__ float rsz_w(int k){
  const float wt[8] = {-0.01171875f,-0.03515625f,0.11328125f,0.43359375f,
                        0.43359375f, 0.11328125f,-0.03515625f,-0.01171875f};
  return wt[k];
}
__global__ void niqe_resizeV_kernel(const float* __restrict__ src, float* __restrict__ dst,
                                    int Hin, int W, int Hout)
{
  int idx = blockIdx.x*blockDim.x + threadIdx.x;
  int total = 8*Hout*W;
  if (idx >= total) return;
  int w = idx % W; int t = idx / W; int i = t % Hout; int b = t / Hout;
  const float* s = src + (size_t)b*Hin*W;
  float acc = 0.f;
  #pragma unroll
  for (int k = 0; k < 8; k++) {
    int p = 2*i + k - 3;
    if (p < 0) p = -p-1; else if (p >= Hin) p = 2*Hin-1-p;   // 'symmetric' pad
    acc += rsz_w(k)*s[(size_t)p*W + w];
  }
  dst[idx] = acc;
}
__global__ void niqe_resizeH_kernel(const float* __restrict__ src, float* __restrict__ dst,
                                    int H, int Win, int Wout)
{
  int idx = blockIdx.x*blockDim.x + threadIdx.x;
  int total = 8*H*Wout;
  if (idx >= total) return;
  int j = idx % Wout; int t = idx / Wout; int h = t % H; int b = t / H;
  const float* s = src + ((size_t)b*H + h)*(size_t)Win;
  float acc = 0.f;
  #pragma unroll
  for (int k = 0; k < 8; k++) {
    int p = 2*j + k - 3;
    if (p < 0) p = -p-1; else if (p >= Win) p = 2*Win-1-p;
    acc += rsz_w(k)*s[p];
  }
  dst[idx] = acc;
}

// ---------------- 4. gamma-ratio lookup table r(gamma), 9801 entries ----------------
__global__ void niqe_rgam_kernel(float* __restrict__ tab)
{
  int i = blockIdx.x*blockDim.x + threadIdx.x;
  if (i >= 9801) return;
  float gm = 0.2f + 0.001f*(float)i;
  tab[i] = expf(2.f*lgammaf(2.f/gm) - lgammaf(1.f/gm) - lgammaf(3.f/gm));
}

// ---------------- 5. per-block AGGD features (18 per block) ----------------
// Block tile is DMA'd into LDS by the Tensor Data Mover (one wave issues the
// descriptor; tile_dim0=tile_dim1=bs, row stride = W, 4-byte elements,
// compacted row-major in LDS), then 8 waves compute the AGGD statistics.
__global__ void niqe_feat_kernel(const float* __restrict__ nrm, int H, int W, int bs,
                                 const float* __restrict__ tab,
                                 float* __restrict__ dist, int scaleOff)
{
  __shared__ float blk[96*96];
  __shared__ float red[256];
  __shared__ int   redi[256];
  __shared__ float stats[6];
  int nbw = W/bs, nbh = H/bs;
  int wid = blockIdx.x;
  int b = wid/(nbh*nbw); int r0 = wid%(nbh*nbw); int bh = r0/nbw, bw = r0%nbw;
  const float* base = nrm + ((size_t)b*H + (size_t)bh*bs)*W + (size_t)bw*bs;
  int tid = threadIdx.x;
  int N = bs*bs;

  if (tid < 32) {   // wave 0 issues the TDM 2D tile load (EXEC-independent, wave-level op)
    unsigned long long ga = (unsigned long long)(const void*)base;   // global byte address
    unsigned int lds_off = (unsigned int)(unsigned long long)(const void*)&blk[0]; // LDS byte offset
    u32x4 g0;
    g0[0] = 1u;                                            // count=1, user descriptor, no gather
    g0[1] = lds_off;                                       // lds_addr
    g0[2] = (unsigned int)(ga & 0xFFFFFFFFu);              // global_addr[31:0]
    g0[3] = (unsigned int)((ga >> 32) & 0x01FFFFFFu) | 0x80000000u; // global_addr[56:32], type=2
    i32x8 g1;
    g1[0] = (int)(2u << 16);                               // data_size=4B; no multicast/pad/iterate
    g1[1] = (int)(((unsigned)W & 0xFFFFu) << 16);          // tensor_dim0[15:0]
    g1[2] = (int)((((unsigned)W >> 16) & 0xFFFFu) | (((unsigned)H & 0xFFFFu) << 16)); // dim0 hi, dim1 lo
    g1[3] = (int)((((unsigned)H >> 16) & 0xFFFFu) | (((unsigned)bs & 0xFFFFu) << 16)); // dim1 hi, tile_dim0
    g1[4] = (int)((unsigned)bs & 0xFFFFu);                 // tile_dim1 (tile_dim2 = 0)
    g1[5] = (int)W;                                        // tensor_dim0_stride[31:0]
    g1[6] = 0;                                             // stride hi / dim1_stride lo
    g1[7] = 0;
    i32x4 z4 = {0,0,0,0};
#if __clang_major__ >= 23
    i32x8 z8 = {0,0,0,0,0,0,0,0};
    __builtin_amdgcn_tensor_load_to_lds(g0, g1, z4, z4, z8, 0);
#else
    __builtin_amdgcn_tensor_load_to_lds(g0, g1, z4, z4, 0);
#endif
    __builtin_amdgcn_s_wait_tensorcnt(0);
  }
  __syncthreads();

  const int sh0[5] = {0,0,1,1,1};
  const int sh1[5] = {0,1,0,1,-1};
  for (int v = 0; v < 5; v++) {
    float cl=0.f, cr=0.f, sl=0.f, sr=0.f, sa=0.f, ss=0.f;
    for (int i = tid; i < N; i += 256) {
      int y = i/bs, x = i%bs;
      float val = blk[i];
      if (v > 0) {        // prod = x * roll(x, sh):  prod[y][x] = x[y][x] * x[(y-s0)%bs][(x-s1)%bs]
        int yy = y - sh0[v]; if (yy < 0) yy += bs;
        int xx = x - sh1[v]; if (xx < 0) xx += bs; else if (xx >= bs) xx -= bs;
        val *= blk[yy*bs + xx];
      }
      float v2 = val*val;
      ss += v2; sa += fabsf(val);
      if (val < 0.f) { cl += 1.f; sl += v2; }
      else if (val > 0.f) { cr += 1.f; sr += v2; }
    }
    float vals[6] = {cl, cr, sl, sr, sa, ss};
    for (int j = 0; j < 6; j++) {
      red[tid] = vals[j]; __syncthreads();
      for (int s = 128; s > 0; s >>= 1) { if (tid < s) red[tid] += red[tid+s]; __syncthreads(); }
      if (tid == 0) stats[j] = red[0];
      __syncthreads();
    }
    float CL=stats[0], CR=stats[1], SL=stats[2], SR=stats[3], SA=stats[4], SS=stats[5];
    float lstd = sqrtf(SL/CL), rstd = sqrtf(SR/CR);
    float gh = lstd/rstd;
    float ma = SA/(float)N, mm2 = SS/(float)N;
    float rhat = ma*ma/mm2;
    float gh2 = gh*gh;
    float rhn = rhat*(gh2*gh + 1.f)*(gh + 1.f)/((gh2 + 1.f)*(gh2 + 1.f));
    // parallel first-min argmin over the 9801-entry table
    float best = 3.4e38f; int bidx = 0x7fffffff;
    for (int t = tid; t < 9801; t += 256) {
      float dlt = fabsf(tab[t] - rhn);
      if (dlt < best) { best = dlt; bidx = t; }
    }
    red[tid] = best; redi[tid] = bidx; __syncthreads();
    for (int s = 128; s > 0; s >>= 1) {
      if (tid < s) {
        if (red[tid+s] < red[tid] || (red[tid+s] == red[tid] && redi[tid+s] < redi[tid])) {
          red[tid] = red[tid+s]; redi[tid] = redi[tid+s];
        }
      }
      __syncthreads();
    }
    if (tid == 0) {
      int pos = redi[0];
      float alpha = 0.2f + 0.001f*(float)pos;
      float bf = sqrtf(expf(lgammaf(1.f/alpha) - lgammaf(3.f/alpha)));
      float bl = lstd*bf, br = rstd*bf;
      float* dp = dist + ((size_t)b*(nbh*nbw) + (size_t)bh*nbw + bw)*36 + scaleOff;
      if (v == 0) { dp[0] = alpha; dp[1] = 0.5f*(bl+br); }
      else {
        float mean = (br - bl)*expf(lgammaf(2.f/alpha) - lgammaf(1.f/alpha));
        int o = 2 + 4*(v-1);
        dp[o] = alpha; dp[o+1] = mean; dp[o+2] = bl; dp[o+3] = br;
      }
    }
    __syncthreads();
  }
}

// ---------------- 6. per-batch mean + centered/padded f16 copy for WMMA ----------------
__global__ void niqe_center_kernel(const float* __restrict__ dist,
                                   float* __restrict__ mu_d, _Float16* __restrict__ xcp)
{
  __shared__ float mu[36];
  int b = blockIdx.x;
  int f = threadIdx.x;
  if (f < 36) {
    float s = 0.f;
    const float* dp = dist + (size_t)b*220*36 + f;
    for (int n = 0; n < 220; n++) s += dp[(size_t)n*36];
    float m = s/220.f;
    mu[f] = m; mu_d[b*36 + f] = m;
  }
  __syncthreads();
  // padded [b][224][48] f16, zeros in padding (contribute 0 to covariance)
  for (int i = threadIdx.x; i < 224*48; i += 64) {
    int n = i/48, ff = i%48;
    float v = 0.f;
    if (n < 220 && ff < 36) v = dist[((size_t)b*220 + n)*36 + ff] - mu[ff];
    xcp[((size_t)b*224 + n)*48 + ff] = (_Float16)v;
  }
}

// ---------------- 7. covariance xc^T xc via V_WMMA_F32_16X16X32_F16, then M=(cov_pris+cov)/2 ----
__global__ void niqe_cov_wmma_kernel(const _Float16* __restrict__ xcp,
                                     const float* __restrict__ cov_pris,
                                     float* __restrict__ M)
{
  // gridDim.x = 8 batches * 9 tiles, blockDim = 32 (one wave per 16x16 output tile)
  int wid = blockIdx.x;
  int b = wid/9; int t = wid%9; int ti = t/3, tj = t%3;
  const _Float16* X = xcp + (size_t)b*224*48;
  int lane = threadIdx.x;
  int half = lane >> 4, lm = lane & 15;
  v8f c = {0.f,0.f,0.f,0.f,0.f,0.f,0.f,0.f};
  #pragma unroll
  for (int kk = 0; kk < 7; kk++) {             // K = 224 in steps of 32, fully unrolled
    v16h a, bm;
    #pragma unroll
    for (int e = 0; e < 16; e++) {
      int vg = e >> 1, p = e & 1;
      // A 16x32 f16 layout: VGPR0..3 -> K={0..7}+8*half, VGPR4..7 -> K={16..23}+8*half
      int ka = (vg < 4) ? (2*vg + p + 8*half) : (16 + 2*(vg-4) + p + 8*half);
      // B 32x16 f16 layout: same K order as A: VGPR0..3 -> K={0..7}+8*half, VGPR4..7 -> K={16..23}+8*half
      int kb = (vg < 4) ? (2*vg + p + 8*half) : (16 + 2*(vg-4) + p + 8*half);
      a[e]  = X[(size_t)(kk*32 + ka)*48 + (ti*16 + lm)];   // A[m][k] = xc[k][m]
      bm[e] = X[(size_t)(kk*32 + kb)*48 + (tj*16 + lm)];   // B[k][n] = xc[k][n]
    }
    c = __builtin_amdgcn_wmma_f32_16x16x32_f16(false, a, false, bm, (short)0, c, false, false);
  }
  #pragma unroll
  for (int r = 0; r < 8; r++) {
    int m = ti*16 + r + 8*half;      // C/D layout: VGPR r -> row r (+8 for upper half-lanes)
    int n = tj*16 + lm;
    if (m < 36 && n < 36) {
      float covv = c[r]/219.f;
      M[((size_t)b*36 + m)*36 + n] = 0.5f*(cov_pris[m*36 + n] + covv);
    }
  }
}

// ---------------- 8. symmetric pinv via cyclic Jacobi + Mahalanobis distance ----------------
__global__ void niqe_pinv_dist_kernel(const float* __restrict__ M,
                                      const float* __restrict__ mu_pris,
                                      const float* __restrict__ mu_d,
                                      float* __restrict__ out)
{
  __shared__ float A[36][36];
  __shared__ float V[36][36];
  __shared__ float lam[36];
  __shared__ float dd[36];
  __shared__ float red[64];
  int b = blockIdx.x;
  int tid = threadIdx.x;               // blockDim = 64
  for (int i = tid; i < 36*36; i += 64) {
    int r = i/36, cn = i%36;
    A[r][cn] = M[(size_t)b*1296 + i];
    V[r][cn] = (r == cn) ? 1.f : 0.f;
  }
  __syncthreads();
  for (int sweep = 0; sweep < 12; sweep++) {
    for (int p = 0; p < 35; p++) {
      for (int q = p+1; q < 36; q++) {
        float apq = A[p][q], app = A[p][p], aqq = A[q][q];
        float cth, sth;
        if (fabsf(apq) > 1e-20f) {
          float tau = (aqq - app)/(2.f*apq);
          float tt = ((tau >= 0.f) ? 1.f : -1.f)/(fabsf(tau) + sqrtf(1.f + tau*tau));
          cth = 1.f/sqrtf(1.f + tt*tt); sth = tt*cth;
        } else { cth = 1.f; sth = 0.f; }
        __syncthreads();
        if (tid < 36) {
          int j = tid;
          float Apj = A[p][j], Aqj = A[q][j];
          A[p][j] = cth*Apj - sth*Aqj;
          A[q][j] = sth*Apj + cth*Aqj;
        }
        __syncthreads();
        if (tid < 36) {
          int j = tid;
          float Ajp = A[j][p], Ajq = A[j][q];
          A[j][p] = cth*Ajp - sth*Ajq;
          A[j][q] = sth*Ajp + cth*Ajq;
          float Vjp = V[j][p], Vjq = V[j][q];
          V[j][p] = cth*Vjp - sth*Vjq;
          V[j][q] = sth*Vjp + cth*Vjq;
        }
        __syncthreads();
      }
    }
  }
  if (tid < 36) lam[tid] = A[tid][tid];
  if (tid < 36) dd[tid] = mu_pris[tid] - mu_d[b*36 + tid];
  __syncthreads();
  float mx = 0.f;
  for (int i = 0; i < 36; i++) mx = fmaxf(mx, fabsf(lam[i]));
  float cutoff = mx * 36.f * 1.1920929e-7f;
  float contrib = 0.f;
  if (tid < 36) {
    float tj = 0.f;
    for (int f2 = 0; f2 < 36; f2++) tj += V[f2][tid]*dd[f2];
    float l = lam[tid];
    float il = (fabsf(l) > cutoff) ? (1.f/l) : 0.f;
    contrib = il*tj*tj;
  }
  red[tid] = contrib; __syncthreads();
  for (int s = 32; s > 0; s >>= 1) { if (tid < s) red[tid] += red[tid+s]; __syncthreads(); }
  if (tid == 0) out[b] = sqrtf(fmaxf(red[0], 0.f));
}

// ---------------------------------- launcher ----------------------------------
extern "C" void kernel_launch(void* const* d_in, const int* in_sizes, int n_in,
                              void* d_out, int out_size, void* d_ws, size_t ws_size,
                              hipStream_t stream) {
  (void)in_sizes; (void)n_in; (void)out_size; (void)ws_size;
  const float* x        = (const float*)d_in[0];  // (8,3,1080,1920)
  const float* mu_pris  = (const float*)d_in[1];  // (1,36)
  const float* cov_pris = (const float*)d_in[2];  // (36,36)
  float* out = (float*)d_out;                     // (8,)
  char* ws = (char*)d_ws;

  auto alignup = [](size_t v){ return (v + 255) & ~((size_t)255); };
  size_t o = 0;
  float*    imgY  = (float*)(ws + o); o = alignup(o + (size_t)16220160*4); // 8x1056x1920
  float*    normB = (float*)(ws + o); o = alignup(o + (size_t)16220160*4); // reused both scales
  float*    tmpV  = (float*)(ws + o); o = alignup(o + (size_t)8110080*4);  // 8x528x1920
  float*    img2  = (float*)(ws + o); o = alignup(o + (size_t)4055040*4);  // 8x528x960
  float*    tab   = (float*)(ws + o); o = alignup(o + (size_t)9801*4);
  float*    dist  = (float*)(ws + o); o = alignup(o + (size_t)63360*4);    // 8x220x36
  float*    mu_d  = (float*)(ws + o); o = alignup(o + (size_t)288*4);
  _Float16* xcp   = (_Float16*)(ws + o); o = alignup(o + (size_t)86016*2); // 8x224x48 f16
  float*    covM  = (float*)(ws + o); o = alignup(o + (size_t)10368*4);    // 8x36x36

  // Y conversion + crop
  niqe_y_kernel<<<DIV_UP(16220160,256),256,0,stream>>>(x, imgY);
  // scale 1: normalize, features
  niqe_musigma_kernel<<<dim3(120,66,8),dim3(16,16),0,stream>>>(imgY, normB, 1056, 1920);
  niqe_rgam_kernel<<<DIV_UP(9801,256),256,0,stream>>>(tab);
  niqe_feat_kernel<<<1760,256,0,stream>>>(normB, 1056, 1920, 96, tab, dist, 0);
  // half-size resize (linear in img; /255 *255 cancels)
  niqe_resizeV_kernel<<<DIV_UP(8110080,256),256,0,stream>>>(imgY, tmpV, 1056, 1920, 528);
  niqe_resizeH_kernel<<<DIV_UP(4055040,256),256,0,stream>>>(tmpV, img2, 528, 1920, 960);
  // scale 2: normalize, features
  niqe_musigma_kernel<<<dim3(60,33,8),dim3(16,16),0,stream>>>(img2, normB, 528, 960);
  niqe_feat_kernel<<<1760,256,0,stream>>>(normB, 528, 960, 48, tab, dist, 18);
  // covariance (WMMA) + pinv + distance
  niqe_center_kernel<<<8,64,0,stream>>>(dist, mu_d, xcp);
  niqe_cov_wmma_kernel<<<72,32,0,stream>>>(xcp, cov_pris, covM);
  niqe_pinv_dist_kernel<<<8,64,0,stream>>>(covM, mu_pris, mu_d, out);
}
